// RetrievalAugmentedWrapper_87033217286718
// MI455X (gfx1250) — matrix operand order repaired and checked by reading, not compiled, not run on hardware
//
#include <hip/hip_runtime.h>
#include <hip/hip_bf16.h>
#include <math.h>

typedef _Float16 v16h __attribute__((ext_vector_type(16)));
typedef float    v8f  __attribute__((ext_vector_type(8)));

#define B_Q     256
#define D_K     512
#define N_DB    100000
#define D_TXT   768
#define C_TS    8
#define T_TS    256
#define TOPK    5
#define GEMM_GRID 256
#define CHUNK_N 128
#define NUM_CHUNKS ((N_DB + CHUNK_N - 1) / CHUNK_N)   // 782
#define NCAND   (GEMM_GRID * TOPK)                    // 1280

// ---------------------------------------------------------------------------
// Top-5 insertion: list sorted descending by value, ties -> lower index first.
// ---------------------------------------------------------------------------
__device__ __forceinline__ void topk_insert(float v, int n, float tv[TOPK], int ti[TOPK]) {
  if (v < tv[TOPK - 1] || (v == tv[TOPK - 1] && n > ti[TOPK - 1])) return;
  int p = TOPK - 1;
  #pragma unroll
  for (int i = TOPK - 2; i >= 0; --i) {
    if (v > tv[i] || (v == tv[i] && n < ti[i])) p = i;   // monotone -> no break needed
  }
  #pragma unroll
  for (int i = TOPK - 1; i > 0; --i) {
    if (i > p) { tv[i] = tv[i - 1]; ti[i] = ti[i - 1]; }
  }
  tv[p] = v; ti[p] = n;
}

// ---------------------------------------------------------------------------
// A-fragment packing offset for V_WMMA_F32_16X16X32_F16 (ISA 7.12.2):
// lane L (m = L%16, half = L/16), element e:
//   k = half*16 + e (contiguous; this code's own order, B uses the same)
// ---------------------------------------------------------------------------
__device__ __forceinline__ int qpack_off(int b, int k) {
  int mt = b >> 4, r = b & 15, kc = k >> 5, kl = k & 31;
  int half, e;
  half = kl >> 4; e = kl & 15;
  int lane = r + (half << 4);
  return ((mt * 16 + kc) * 32 + lane) * 16 + e;
}

// ---------------------------------------------------------------------------
// Phase 0: L2-normalize queries, convert to f16, scatter into A-frag layout.
// grid = 256 (one query per block), block = 256
// ---------------------------------------------------------------------------
__global__ __launch_bounds__(256) void qnorm_pack_kernel(
    const float* __restrict__ xq, _Float16* __restrict__ qpack)
{
  __shared__ float red[256];
  const int b = blockIdx.x, t = threadIdx.x;
  float x0 = xq[(size_t)b * D_K + t];
  float x1 = xq[(size_t)b * D_K + 256 + t];
  red[t] = x0 * x0 + x1 * x1;
  __syncthreads();
  for (int s = 128; s > 0; s >>= 1) {
    if (t < s) red[t] += red[t + s];
    __syncthreads();
  }
  float scale = 1.0f / fmaxf(sqrtf(red[0]), 1e-12f);
  qpack[qpack_off(b, t)]       = (_Float16)(x0 * scale);
  qpack[qpack_off(b, 256 + t)] = (_Float16)(x1 * scale);
}

// ---------------------------------------------------------------------------
// Phase 1: fused WMMA GEMM + per-block top-5.
// grid = 256 blocks x 256 threads (8 waves). Each chunk = 128 bank rows.
// Each wave owns 16 columns; 16 accumulators cover all 256 queries so
// ts_emb is streamed from HBM exactly once.
// B-frag k order chosen to match qpack (dense 16x16x32 f16 B, KxN): lane L holds column
// n = L%16; elements e=0..15 hold K = (L/16)*16 + e  (contiguous 64B read).
// ---------------------------------------------------------------------------
__global__ __launch_bounds__(256) void sim_topk_kernel(
    const float* __restrict__ ts_emb, const _Float16* __restrict__ qpack,
    float* __restrict__ cand_val, int* __restrict__ cand_idx)
{
  __shared__ float SIM[B_Q * CHUNK_N];   // 128 KB, columns rotated by row ((c + q) & 127)

  const int tid  = threadIdx.x;
  const int lane = tid & 31;
  const int wv   = tid >> 5;     // wave 0..7
  const int l16  = lane & 15;
  const int lh   = lane >> 4;    // half-wave 0/1

  float tv[TOPK]; int ti[TOPK];
  #pragma unroll
  for (int j = 0; j < TOPK; ++j) { tv[j] = -3.4e38f; ti[j] = 0x7fffffff; }

  for (int chunk = blockIdx.x; chunk < NUM_CHUNKS; chunk += GEMM_GRID) {
    const int nbase = chunk * CHUNK_N;
    const int col   = nbase + wv * 16 + l16;
    const int rowc  = (col < N_DB) ? col : 0;            // clamp; masked later
    const float* brow = ts_emb + (size_t)rowc * D_K + (lh << 4);
    __builtin_prefetch(brow, 0, 0);                      // global_prefetch_b8

    v8f acc[16];
    #pragma unroll
    for (int mt = 0; mt < 16; ++mt)
      #pragma unroll
      for (int e = 0; e < 8; ++e) acc[mt][e] = 0.0f;

    for (int kc = 0; kc < 16; ++kc) {
      const float4* bp = (const float4*)(brow + kc * 32);
      float4 f0 = bp[0], f1 = bp[1], f2 = bp[2], f3 = bp[3];
      v16h bf;
      bf[0]  = (_Float16)f0.x; bf[1]  = (_Float16)f0.y;
      bf[2]  = (_Float16)f0.z; bf[3]  = (_Float16)f0.w;
      bf[4]  = (_Float16)f1.x; bf[5]  = (_Float16)f1.y;
      bf[6]  = (_Float16)f1.z; bf[7]  = (_Float16)f1.w;
      bf[8]  = (_Float16)f2.x; bf[9]  = (_Float16)f2.y;
      bf[10] = (_Float16)f2.z; bf[11] = (_Float16)f2.w;
      bf[12] = (_Float16)f3.x; bf[13] = (_Float16)f3.y;
      bf[14] = (_Float16)f3.z; bf[15] = (_Float16)f3.w;

      #pragma unroll
      for (int mt = 0; mt < 16; ++mt) {
        v16h af = *(const v16h*)(qpack + ((size_t)((mt * 16 + kc) * 32 + lane) << 4));
        acc[mt] = __builtin_amdgcn_wmma_f32_16x16x32_f16(
            false, af, false, bf, (short)0, acc[mt], false, false);
      }
    }

    // C/D layout: lane L, VGPR r -> (m = r + 8*(L/16), n = L%16)
    #pragma unroll
    for (int mt = 0; mt < 16; ++mt) {
      const int qbase = mt * 16 + lh * 8;
      const int c     = wv * 16 + l16;
      #pragma unroll
      for (int r = 0; r < 8; ++r) {
        const int q = qbase + r;
        SIM[(q << 7) | ((c + q) & 127)] = acc[mt][r];
      }
    }
    __syncthreads();

    // thread t scans query t's 128 columns
    for (int c = 0; c < CHUNK_N; ++c) {
      const int n = nbase + c;
      const float v = SIM[(tid << 7) | ((c + tid) & 127)];
      if (n < N_DB) topk_insert(v, n, tv, ti);
    }
    __syncthreads();
  }

  #pragma unroll
  for (int j = 0; j < TOPK; ++j) {
    const size_t o = ((size_t)tid * GEMM_GRID + blockIdx.x) * TOPK + j;
    cand_val[o] = tv[j];
    cand_idx[o] = ti[j];
  }
}

// ---------------------------------------------------------------------------
// Phase 2: merge 1280 candidates per query -> final top-5 (descending).
// grid = 256 (one query per block), block = 256
// ---------------------------------------------------------------------------
__global__ __launch_bounds__(256) void topk_reduce_kernel(
    const float* __restrict__ cv, const int* __restrict__ ci, int* __restrict__ fidx)
{
  __shared__ float sv[NCAND];
  __shared__ int   si[NCAND];
  __shared__ float mv[16 * TOPK];
  __shared__ int   mi[16 * TOPK];

  const int q = blockIdx.x, t = threadIdx.x;
  for (int i = t; i < NCAND; i += 256) {
    sv[i] = cv[(size_t)q * NCAND + i];
    si[i] = ci[(size_t)q * NCAND + i];
  }
  __syncthreads();

  if (t < 16) {
    float tv[TOPK]; int ti[TOPK];
    #pragma unroll
    for (int j = 0; j < TOPK; ++j) { tv[j] = -3.4e38f; ti[j] = 0x7fffffff; }
    const int seg = NCAND / 16;   // 80
    for (int j = 0; j < seg; ++j) {
      const int k = t * seg + j;
      topk_insert(sv[k], si[k], tv, ti);
    }
    #pragma unroll
    for (int j = 0; j < TOPK; ++j) { mv[t * TOPK + j] = tv[j]; mi[t * TOPK + j] = ti[j]; }
  }
  __syncthreads();

  if (t == 0) {
    float tv[TOPK]; int ti[TOPK];
    #pragma unroll
    for (int j = 0; j < TOPK; ++j) { tv[j] = -3.4e38f; ti[j] = 0x7fffffff; }
    for (int k = 0; k < 16 * TOPK; ++k) topk_insert(mv[k], mi[k], tv, ti);
    #pragma unroll
    for (int j = 0; j < TOPK; ++j) fidx[(size_t)q * TOPK + j] = ti[j];
  }
}

// ---------------------------------------------------------------------------
// Phase 3: gather rows into concatenated output.
// out = [text_topk 256*5*768 | ts_topk 256*5*512 | timeseries_topk 256*5*2048]
// grid = 1280 (one (b,k) pair per block), block = 256
// ---------------------------------------------------------------------------
__global__ __launch_bounds__(256) void gather_kernel(
    const float* __restrict__ text_emb, const float* __restrict__ ts_emb,
    const float* __restrict__ tseries, const int* __restrict__ fidx,
    float* __restrict__ out)
{
  const int pair = blockIdx.x;         // 0..1279
  const int t    = threadIdx.x;
  const int idx  = fidx[pair];

  const float4* s1 = (const float4*)(text_emb + (size_t)idx * D_TXT);
  float4*       o1 = (float4*)(out + (size_t)pair * D_TXT);
  for (int i = t; i < D_TXT / 4; i += 256) o1[i] = s1[i];

  const float4* s2 = (const float4*)(ts_emb + (size_t)idx * D_K);
  float4*       o2 = (float4*)(out + (size_t)B_Q * TOPK * D_TXT + (size_t)pair * D_K);
  for (int i = t; i < D_K / 4; i += 256) o2[i] = s2[i];

  const float4* s3 = (const float4*)(tseries + (size_t)idx * (C_TS * T_TS));
  float4*       o3 = (float4*)(out + (size_t)B_Q * TOPK * (D_TXT + D_K)
                                   + (size_t)pair * (C_TS * T_TS));
  for (int i = t; i < (C_TS * T_TS) / 4; i += 256) o3[i] = s3[i];
}

// ---------------------------------------------------------------------------
extern "C" void kernel_launch(void* const* d_in, const int* in_sizes, int n_in,
                              void* d_out, int out_size, void* d_ws, size_t ws_size,
                              hipStream_t stream) {
  const float* x_query    = (const float*)d_in[0];   // [256, 512]
  const float* ts_emb     = (const float*)d_in[1];   // [100000, 512]
  const float* text_emb   = (const float*)d_in[2];   // [100000, 768]
  const float* timeseries = (const float*)d_in[3];   // [100000, 8, 256]
  // d_in[4] = top_k (5), compile-time constant here.
  float* out = (float*)d_out;

  char* w = (char*)d_ws;
  const size_t QPACK_BYTES = (size_t)B_Q * D_K * sizeof(_Float16);       // 262144
  const size_t CAND_BYTES  = (size_t)B_Q * NCAND * sizeof(float);        // 1310720
  _Float16* qpack    = (_Float16*)(w);
  float*    cand_val = (float*)(w + QPACK_BYTES);
  int*      cand_idx = (int*)(w + QPACK_BYTES + CAND_BYTES);
  int*      fidx     = (int*)(w + QPACK_BYTES + 2 * CAND_BYTES);

  qnorm_pack_kernel<<<B_Q, 256, 0, stream>>>(x_query, qpack);
  sim_topk_kernel<<<GEMM_GRID, 256, 0, stream>>>(ts_emb, qpack, cand_val, cand_idx);
  topk_reduce_kernel<<<B_Q, 256, 0, stream>>>(cand_val, cand_idx, fidx);
  gather_kernel<<<B_Q * TOPK, 256, 0, stream>>>(text_emb, ts_emb, timeseries, fidx, out);
}
